// AttentionLayer_31413390803349
// MI455X (gfx1250) — matrix operand order repaired and checked by reading, not compiled, not run on hardware
//
#include <hip/hip_runtime.h>
#include <hip/hip_bf16.h>
#include <cstdint>

// Problem constants (match reference)
#define Ecnt 1024
#define Hcnt 1024
#define Bcnt 32
#define Lcnt 2048
#define LT   32    // L-columns per workgroup tile

typedef __attribute__((ext_vector_type(16))) _Float16 v16h;
typedef __attribute__((ext_vector_type(4)))  _Float16 v4h;
typedef __attribute__((ext_vector_type(8)))  float    v8f;
typedef __attribute__((ext_vector_type(4)))  unsigned int v4u;
typedef __attribute__((ext_vector_type(8)))  int      v8i;
typedef __attribute__((ext_vector_type(4)))  int      v4i;

#define WMMA_F16(A, B, C) __builtin_amdgcn_wmma_f32_16x16x32_f16(false, (A), false, (B), (short)0, (C), false, false)

// ---------------------------------------------------------------------------
// Pre-swizzle W (f32 [1024,1024] row-major) into WMMA A-matrix 16x32 f16 tiles.
// Tile (mt, kt) stored as [lane 0..31][elem 0..15] contiguous (512 halfs).
// Contiguous k order (not the ISA table order): lane l holds row M = mt*16 + (l&15); element j holds
//   K = kt*32 + (l>=16 ? 16 : 0) + j.
// ---------------------------------------------------------------------------
__global__ void convert_w(const float* __restrict__ W, _Float16* __restrict__ Wt) {
    int gid  = blockIdx.x * 256 + threadIdx.x;      // 0 .. 1M-1
    int j    = gid & 15;
    int lane = (gid >> 4) & 31;
    int tile = gid >> 9;
    int kt   = tile & 31;                            // E/32 = 32 k-tiles
    int mt   = tile >> 5;
    int M = mt * 16 + (lane & 15);
    int K = kt * 32 + ((lane >> 4) ? 16 : 0) + j;
    Wt[gid] = (_Float16)W[(size_t)M * 1024 + K];
}

__global__ void zero_f32(float* __restrict__ p, int n) {
    int i = blockIdx.x * 256 + threadIdx.x;
    if (i < n) p[i] = 0.f;
}

// ---------------------------------------------------------------------------
// Small dense layers: out[b,g] = act(bias[g] + sum_k (in[b,k] (+in2[b,k])) * W[g,k])
// One wave (32 lanes) per output element; 8 outputs per 256-thread block.
// ---------------------------------------------------------------------------
__global__ void small_gemm(const float* __restrict__ in, const float* __restrict__ in2,
                           const float* __restrict__ W, const float* __restrict__ bias,
                           float* __restrict__ out, int doTanh) {
    int wv   = threadIdx.x >> 5;
    int lane = threadIdx.x & 31;
    int idx  = blockIdx.x * 8 + wv;                  // 0 .. B*H-1
    int b = idx >> 10, g = idx & 1023;
    const float* ip = in + (size_t)b * 1024;
    const float* wp = W  + (size_t)g * 1024;
    float s = 0.f;
    if (in2) {
        const float* i2 = in2 + (size_t)b * 1024;
        for (int k = lane; k < 1024; k += 32) s += (ip[k] + i2[k]) * wp[k];
    } else {
        for (int k = lane; k < 1024; k += 32) s += ip[k] * wp[k];
    }
#pragma unroll
    for (int off = 16; off; off >>= 1) s += __shfl_xor(s, off, 32);
    if (lane == 0) {
        float r = s + bias[g];
        out[idx] = doTanh ? tanhf(r) : r;
    }
}

// ---------------------------------------------------------------------------
// Fused main kernel: per (batch b, L-tile of 32 columns):
//   TDM async-loads the raw f32 feature tile (1024 rows x 128B @ 8KB stride)
//   f  = relu(W_lf @ feat + b_lf)            (WMMA GEMM1, H x 32)
//   fe = W_ef @ f + b_ef                     (WMMA GEMM2, H x 32)
//   scores[:, tile] += w_att . tanh(fe + he) (atomic f32)
//   f TDM async-stored to workspace (f16 B-layout) for the attention sum.
// 8 waves; wave w owns output rows [128w, 128w+128) = 8 m-tiles x 2 n-tiles.
// LDS map (dynamic, 208896 B):
//   [0, 131072)        sRaw  raw f32 feature tile (dead after staging)
//   [0,  65536)        sF    f16 f tile  (aliases sRaw; written in phase-1 epi)
//   [131072, 196608)   sFeat f16 feature tile, WMMA B-layout
//   [196608, 208896)   b_lf | (b_ef+he) | w_att  (f32 x 1024 each)
// ---------------------------------------------------------------------------
__global__ void __launch_bounds__(256)
fused_main(const float* __restrict__ features,
           const _Float16* __restrict__ Wlf16,
           const _Float16* __restrict__ Wef16,
           const float* __restrict__ b_lf,
           const float* __restrict__ b_ef,
           const float* __restrict__ he,
           const float* __restrict__ w_att,
           float* __restrict__ scores,
           _Float16* __restrict__ fOut) {
    extern __shared__ __align__(32) char smem[];
    float*    sRaw  = (float*)smem;                  // 32x1024 f32 (128KB)
    _Float16* sF    = (_Float16*)smem;               // aliases sRaw (64KB)
    _Float16* sFeat = (_Float16*)(smem + 131072);    // 32768 halfs (64KB)
    float*    sBlf  = (float*)(smem + 196608);
    float*    sBhe  = (float*)(smem + 200704);
    float*    sWatt = (float*)(smem + 204800);

    const int t     = threadIdx.x;
    const int ltile = blockIdx.x;
    const int b     = blockIdx.y;
    const int l0    = ltile * LT;

    // ---- Phase 0a: TDM async DMA of the strided feature tile -> sRaw -------
    // 2D tile: tile_dim0 = 32 elems (4B), tile_dim1 = 1024 rows, row stride
    // tensor_dim0_stride = L = 2048 elems. One instruction per workgroup.
    if (t < 32) {   // wave 0 only (tensor ops ignore EXEC; issue once)
        uint64_t ga = (uint64_t)(uintptr_t)(features + ((size_t)b * Ecnt) * Lcnt + l0);
        v4u g0;
        g0.x = 1u;                                        // count=1, user mode
        g0.y = (unsigned)(uintptr_t)sRaw;                 // lds_addr (low 32 = LDS offset)
        g0.z = (unsigned)(ga & 0xFFFFFFFFu);              // global_addr[31:0]
        g0.w = (unsigned)((ga >> 32) & 0x1FFFFFFu) | (2u << 30);  // addr[56:32] | type=2
        v8i g1;
        g1[0] = (2 << 16);                // data_size = 4B
        g1[1] = (int)(32u << 16);         // tensor_dim0 = 32
        g1[2] = (int)(1024u << 16);       // tensor_dim1 = 1024
        g1[3] = (int)(32u << 16);         // tile_dim0 = 32
        g1[4] = 1024;                     // tile_dim1 = 1024 (tile_dim2 = 0 -> 2D)
        g1[5] = 2048;                     // tensor_dim0_stride = L
        g1[6] = 0;
        g1[7] = 0;
        v4i gz = {};                      // groups 2/3 unused (2D tensor)
        v8i gz8 = {};                     // extra operand of the 6-arg builtin
        __builtin_amdgcn_tensor_load_to_lds(g0, g1, gz, gz, gz8, 0);
        __builtin_amdgcn_s_wait_tensorcnt(0);
    }

    // ---- Phase 0b: bias vectors (overlaps with nothing; cheap) -------------
    for (int i = t; i < 1024; i += 256) {
        sBlf[i]  = b_lf[i];
        sBhe[i]  = b_ef[i] + he[(size_t)b * Hcnt + i];
        sWatt[i] = w_att[i];
    }
    __syncthreads();   // sRaw complete + visible

    // ---- Phase 0c: swizzle raw f32 tile -> f16 WMMA B-layout in sFeat ------
    {
        const int er = t >> 3;            // row within 32-row group
        const int c0 = (t & 7) * 4;       // 4 consecutive columns
        for (int e0 = 0; e0 < Ecnt; e0 += 32) {
            int e = e0 + er;
            const float4 v = *(const float4*)(sRaw + (size_t)e * 32 + c0);
            float vv[4] = {v.x, v.y, v.z, v.w};
#pragma unroll
            for (int u = 0; u < 4; ++u) {
                int n = c0 + u;
                // B-layout: lane = 16*((e%32)/16) + (n%16), elem j = e%16
                int idx = (((e >> 5) * 2 + (n >> 4)) * 32 + (((e >> 4) & 1) * 16 + (n & 15))) * 16 + (e & 15);
                sFeat[idx] = (_Float16)vv[u];
            }
        }
    }
    __syncthreads();   // sFeat ready; sRaw now dead (sF may overwrite)

    const int w      = t >> 5;
    const int lane   = t & 31;
    const int laneHi = lane >> 4;
    const int laneLo = lane & 15;

    // ---- Phase 1: f = relu(W_lf @ feat + b_lf) ----
    {
        v8f C1[8][2] = {};
        const _Float16* wbase = Wlf16 + ((size_t)(w * 8) * 32) * 512 + lane * 16;
        // 2-deep A-tile ring (two global loads in flight) + B double buffer
        v16h Ar0 = *(const v16h*)(wbase + (size_t)(0 * 32 + 0) * 512);
        v16h Ar1 = *(const v16h*)(wbase + (size_t)(1 * 32 + 0) * 512);
        v16h Bc0 = *(const v16h*)(sFeat + (0 * 2 + 0) * 512 + lane * 16);
        v16h Bc1 = *(const v16h*)(sFeat + (0 * 2 + 1) * 512 + lane * 16);
#pragma unroll 1
        for (int kt = 0; kt < 32; ++kt) {
            int ktn = (kt < 31) ? kt + 1 : 31;           // B prefetch (clamped)
            v16h Bn0 = *(const v16h*)(sFeat + (ktn * 2 + 0) * 512 + lane * 16);
            v16h Bn1 = *(const v16h*)(sFeat + (ktn * 2 + 1) * 512 + lane * 16);
#pragma unroll
            for (int m = 0; m < 8; ++m) {
                v16h A = (m & 1) ? Ar1 : Ar0;
                // prefetch A tile two iterations ahead (over-read at end is
                // in-bounds of the 2048-tile array and never consumed)
                int pm  = (m < 6) ? (m + 2) : (m - 6);
                int pkt = (m < 6) ? kt : (kt + 1);
                v16h An = *(const v16h*)(wbase + (size_t)(pm * 32 + pkt) * 512);
                if (m & 1) Ar1 = An; else Ar0 = An;
                C1[m][0] = WMMA_F16(A, Bc0, C1[m][0]);
                C1[m][1] = WMMA_F16(A, Bc1, C1[m][1]);
            }
            Bc0 = Bn0; Bc1 = Bn1;
        }
        // epilogue: bias + relu, store f into LDS (B-layout) for GEMM2
#pragma unroll
        for (int m = 0; m < 8; ++m) {
            const int mtAbs = w * 8 + m;
#pragma unroll
            for (int nt = 0; nt < 2; ++nt) {
#pragma unroll
                for (int r = 0; r < 8; ++r) {
                    int M = mtAbs * 16 + laneHi * 8 + r;     // C/D layout: M = r + 8*(lane/16)
                    float val = C1[m][nt][r] + sBlf[M];
                    val = val > 0.f ? val : 0.f;
                    int idx = (((M >> 5) * 2 + nt) * 32 + (((M >> 4) & 1) * 16 + laneLo)) * 16 + (M & 15);
                    sF[idx] = (_Float16)val;
                }
            }
        }
    }
    __syncthreads();

    // ---- TDM async store: f tile (contiguous 64KB) -> workspace ------------
    // Fire-and-forget: only *reads* sF (phase 2 also only reads it);
    // S_ENDPGM's implicit wait-idle drains TENSORcnt.
    if (t < 32) {
        uint64_t ga = (uint64_t)(uintptr_t)(fOut + ((size_t)(b * 64 + ltile)) * 32768);
        v4u g0;
        g0.x = 1u;
        g0.y = (unsigned)(uintptr_t)sF;
        g0.z = (unsigned)(ga & 0xFFFFFFFFu);
        g0.w = (unsigned)((ga >> 32) & 0x1FFFFFFu) | (2u << 30);
        v8i g1;
        g1[0] = (1 << 16);                // data_size = 2B
        g1[1] = (int)(32768u << 16);      // tensor_dim0 = 32768
        g1[2] = (int)(1u << 16);          // tensor_dim1 = 1
        g1[3] = (int)(32768u << 16);      // tile_dim0 = 32768
        g1[4] = 1;                        // tile_dim1 = 1
        g1[5] = 32768;                    // tensor_dim0_stride
        g1[6] = 0;
        g1[7] = 0;
        v4i gz = {};
        v8i gz8 = {};
        __builtin_amdgcn_tensor_store_from_lds(g0, g1, gz, gz, gz8, 0);
    }

    // ---- Phase 2: fe = W_ef @ f + b_ef; scores += w_att . tanh(fe + he) ----
    {
        v8f C2[8][2] = {};
        const _Float16* wbase = Wef16 + ((size_t)(w * 8) * 32) * 512 + lane * 16;
        v16h Ar0 = *(const v16h*)(wbase + (size_t)(0 * 32 + 0) * 512);
        v16h Ar1 = *(const v16h*)(wbase + (size_t)(1 * 32 + 0) * 512);
        v16h Bc0 = *(const v16h*)(sF + (0 * 2 + 0) * 512 + lane * 16);
        v16h Bc1 = *(const v16h*)(sF + (0 * 2 + 1) * 512 + lane * 16);
#pragma unroll 1
        for (int kt = 0; kt < 32; ++kt) {
            int ktn = (kt < 31) ? kt + 1 : 31;
            v16h Bn0 = *(const v16h*)(sF + (ktn * 2 + 0) * 512 + lane * 16);
            v16h Bn1 = *(const v16h*)(sF + (ktn * 2 + 1) * 512 + lane * 16);
#pragma unroll
            for (int m = 0; m < 8; ++m) {
                v16h A = (m & 1) ? Ar1 : Ar0;
                int pm  = (m < 6) ? (m + 2) : (m - 6);
                int pkt = (m < 6) ? kt : (kt + 1);
                v16h An = *(const v16h*)(wbase + (size_t)(pm * 32 + pkt) * 512);
                if (m & 1) Ar1 = An; else Ar0 = An;
                C2[m][0] = WMMA_F16(A, Bc0, C2[m][0]);
                C2[m][1] = WMMA_F16(A, Bc1, C2[m][1]);
            }
            Bc0 = Bn0; Bc1 = Bn1;
        }
        float sc0 = 0.f, sc1 = 0.f;
#pragma unroll
        for (int m = 0; m < 8; ++m) {
            const int mtAbs = w * 8 + m;
#pragma unroll
            for (int r = 0; r < 8; ++r) {
                int M = mtAbs * 16 + laneHi * 8 + r;
                float wa = sWatt[M], bh = sBhe[M];
                sc0 += wa * tanhf(C2[m][0][r] + bh);
                sc1 += wa * tanhf(C2[m][1][r] + bh);
            }
        }
        atomicAdd(scores + (size_t)b * Lcnt + l0 + laneLo,      sc0);
        atomicAdd(scores + (size_t)b * Lcnt + l0 + 16 + laneLo, sc1);
    }
}

// ---------------------------------------------------------------------------
// Row softmax over L=2048 (b_att omitted: softmax is shift-invariant, and
// neither returned output depends on un-normalized scores).
// ---------------------------------------------------------------------------
__global__ void softmax_rows(const float* __restrict__ scores, float* __restrict__ att) {
    __shared__ float srow[2048];
    __shared__ float sred[256];
    int b = blockIdx.x, t = threadIdx.x;
    const float* row = scores + (size_t)b * 2048;
    float mx = -3.4e38f;
    for (int i = t; i < 2048; i += 256) { float v = row[i]; srow[i] = v; mx = fmaxf(mx, v); }
    sred[t] = mx; __syncthreads();
    for (int s = 128; s > 0; s >>= 1) { if (t < s) sred[t] = fmaxf(sred[t], sred[t + s]); __syncthreads(); }
    mx = sred[0]; __syncthreads();
    float sum = 0.f;
    for (int i = t; i < 2048; i += 256) { float e = expf(srow[i] - mx); srow[i] = e; sum += e; }
    sred[t] = sum; __syncthreads();
    for (int s = 128; s > 0; s >>= 1) { if (t < s) sred[t] += sred[t + s]; __syncthreads(); }
    float inv = 1.f / sred[0];
    for (int i = t; i < 2048; i += 256) att[(size_t)b * 2048 + i] = srow[i] * inv;
}

// ---------------------------------------------------------------------------
// v[b,h] += sum_{l in tile} att[b,l] * f[b,h,l]   (f stored in B-layout tiles)
// grid (L/32, B); thread owns 4 consecutive h -> one 8-byte f16x4 load per l.
// ---------------------------------------------------------------------------
__global__ void weighted_sum(const _Float16* __restrict__ fOut, const float* __restrict__ att,
                             float* __restrict__ vacc) {
    __shared__ float sAtt[32];
    int ltile = blockIdx.x, b = blockIdx.y, t = threadIdx.x;
    if (t < 32) sAtt[t] = att[(size_t)b * 2048 + ltile * 32 + t];
    __syncthreads();
    const _Float16* chunk = fOut + ((size_t)(b * 64 + ltile)) * 32768;
    int h0       = t * 4;
    int kt       = h0 >> 5;
    int laneHalf = (h0 >> 4) & 1;
    int j        = h0 & 15;
    float a0 = 0.f, a1 = 0.f, a2 = 0.f, a3 = 0.f;
#pragma unroll
    for (int lc = 0; lc < 32; ++lc) {
        int nt    = lc >> 4;
        int lane2 = laneHalf * 16 + (lc & 15);
        v4h f4 = *(const v4h*)(chunk + (((kt * 2 + nt) * 32 + lane2) * 16 + j));
        float a = sAtt[lc];
        a0 += a * (float)f4.x; a1 += a * (float)f4.y;
        a2 += a * (float)f4.z; a3 += a * (float)f4.w;
    }
    float* vp = vacc + (size_t)b * 1024 + h0;
    atomicAdd(vp + 0, a0); atomicAdd(vp + 1, a1);
    atomicAdd(vp + 2, a2); atomicAdd(vp + 3, a3);
}

// ---------------------------------------------------------------------------
extern "C" void kernel_launch(void* const* d_in, const int* in_sizes, int n_in,
                              void* d_out, int out_size, void* d_ws, size_t ws_size,
                              hipStream_t stream) {
    (void)in_sizes; (void)n_in; (void)out_size;
    const float* features = (const float*)d_in[0];
    const float* hidden   = (const float*)d_in[1];
    const float* W_lf     = (const float*)d_in[2];
    const float* b_lf     = (const float*)d_in[3];
    const float* W_ef     = (const float*)d_in[4];
    const float* b_ef     = (const float*)d_in[5];
    const float* W_lh     = (const float*)d_in[6];
    const float* b_lh     = (const float*)d_in[7];
    const float* W_eh     = (const float*)d_in[8];
    const float* b_eh     = (const float*)d_in[9];
    const float* w_att    = (const float*)d_in[10];
    // d_in[11] = b_att: unused (softmax shift-invariance)
    const float* W_out    = (const float*)d_in[12];
    const float* b_out    = (const float*)d_in[13];

    float* out = (float*)d_out;          // [B,H] = 32768 f32
    float* att = out + Bcnt * Hcnt;      // [B,L] = 65536 f32

    // Workspace layout (bytes); total ~132.6 MB
    char* ws = (char*)d_ws;
    _Float16* Wlf16  = (_Float16*)(ws + 0);                 // 2 MB
    _Float16* Wef16  = (_Float16*)(ws + 2097152);           // 2 MB
    float*    h_act  = (float*)(ws + 4194304);              // 128 KB
    float*    he_buf = (float*)(ws + 4325376);              // 128 KB
    float*    sc_buf = (float*)(ws + 4456448);              // 256 KB
    float*    v_buf  = (float*)(ws + 4718592);              // 128 KB
    _Float16* f_buf  = (_Float16*)(ws + 4849664);           // 128 MB
    (void)ws_size;  // requires ~139 MB of scratch

    // 1) pre-swizzle weights to f16 A-layout
    convert_w<<<4096, 256, 0, stream>>>(W_lf, Wlf16);
    convert_w<<<4096, 256, 0, stream>>>(W_ef, Wef16);

    // 2) h = tanh(hidden @ W_lh^T + b_lh); he = h @ W_eh^T + b_eh
    small_gemm<<<4096, 256, 0, stream>>>(hidden, nullptr, W_lh, b_lh, h_act, 1);
    small_gemm<<<4096, 256, 0, stream>>>(h_act,  nullptr, W_eh, b_eh, he_buf, 0);

    // 3) zero accumulators (scores, v)
    zero_f32<<<(Bcnt * Lcnt + 255) / 256, 256, 0, stream>>>(sc_buf, Bcnt * Lcnt);
    zero_f32<<<(Bcnt * Hcnt + 255) / 256, 256, 0, stream>>>(v_buf, Bcnt * Hcnt);

    // 4) fused WMMA GEMM1 + GEMM2 + score epilogue (TDM staged)
    fused_main<<<dim3(Lcnt / LT, Bcnt), 256, 208896, stream>>>(
        features, Wlf16, Wef16, b_lf, b_ef, he_buf, w_att, sc_buf, f_buf);

    // 5) att = softmax(scores) -> written straight into d_out's att slot
    softmax_rows<<<Bcnt, 256, 0, stream>>>(sc_buf, att);

    // 6) v = einsum('bl,bhl->bh', att, f)
    weighted_sum<<<dim3(Lcnt / LT, Bcnt), 256, 0, stream>>>(f_buf, att, v_buf);

    // 7) out = tanh((v + h) @ W_out^T + b_out)
    small_gemm<<<4096, 256, 0, stream>>>(v_buf, h_act, W_out, b_out, out, 1);
}
